// EncoderLayer_75299366634050
// MI455X (gfx1250) — matrix operand order repaired and checked by reading, not compiled, not run on hardware
//
#include <hip/hip_runtime.h>
#include <hip/hip_bf16.h>
#include <math.h>

// ---------------------------------------------------------------------------
// Types matching gfx1250 WMMA builtins
// ---------------------------------------------------------------------------
typedef __attribute__((ext_vector_type(16))) _Float16 v16h;
typedef __attribute__((ext_vector_type(8)))  _Float16 v8h;
typedef __attribute__((ext_vector_type(4)))  _Float16 v4h;
typedef __attribute__((ext_vector_type(2)))  _Float16 v2h;
typedef __attribute__((ext_vector_type(8)))  float    v8f;

// b128 payload type exactly as the async-LDS builtin expects (generic vector).
typedef int b128i __attribute__((vector_size(16)));

__device__ __forceinline__ v16h cat16(v8h a, v8h b) {
    return __builtin_shufflevector(a, b, 0, 1, 2, 3, 4, 5, 6, 7,
                                   8, 9, 10, 11, 12, 13, 14, 15);
}

// Pack a float4 into 4 halves with v_cvt_pk_f16_f32.
__device__ __forceinline__ v4h pk4(float4 v) {
    v2h a = __builtin_bit_cast(v2h, __builtin_amdgcn_cvt_pkrtz(v.x, v.y));
    v2h b = __builtin_bit_cast(v2h, __builtin_amdgcn_cvt_pkrtz(v.z, v.w));
    return __builtin_shufflevector(a, b, 0, 1, 2, 3);
}

// ---------------------------------------------------------------------------
// Async global->LDS copy (gfx1250 GLOBAL_LOAD_ASYNC_TO_LDS_B128, ASYNCcnt)
// with safe fallback to a plain b128 copy if the builtin is unavailable.
// Builtin signature: (v4i addrspace(1)*, v4i addrspace(3)*, imm off, imm cpol)
// ---------------------------------------------------------------------------
#if defined(__has_builtin)
#if __has_builtin(__builtin_amdgcn_global_load_async_to_lds_b128) && \
    __has_builtin(__builtin_amdgcn_s_wait_asynccnt)
#define ASYNC_LDS 1
#endif
#endif

__device__ __forceinline__ void copy16_g2l(const _Float16* g, _Float16* l) {
#ifdef ASYNC_LDS
    __builtin_amdgcn_global_load_async_to_lds_b128(
        (__attribute__((address_space(1))) b128i*)(uintptr_t)g,
        (__attribute__((address_space(3))) b128i*)(uintptr_t)(unsigned)(uintptr_t)l,
        0, 0);
#else
    *(v8h*)l = *(const v8h*)g;
#endif
}

__device__ __forceinline__ void async_wait() {
#ifdef ASYNC_LDS
    __builtin_amdgcn_s_wait_asynccnt(0);
#endif
}

// ---------------------------------------------------------------------------
// fp32 -> f16 elementwise convert (for the layer input x).
// ---------------------------------------------------------------------------
__global__ __launch_bounds__(256)
void f32to16_kernel(const float* __restrict__ X, _Float16* __restrict__ Y)
{
    const size_t i = ((size_t)blockIdx.x * 256 + threadIdx.x) * 4;
    const float4 v = *(const float4*)(X + i);
    *(v4h*)(Y + i) = pk4(v);
}

// ---------------------------------------------------------------------------
// Weight transpose + fp32 -> f16 convert:  Wt[n][k] = (f16)W[k][n]
// grid = (N/64, K/64), block = 256.
// ---------------------------------------------------------------------------
__global__ __launch_bounds__(256)
void wtrans_kernel(const float* __restrict__ W, _Float16* __restrict__ Wt,
                   int K, int N)
{
    __shared__ _Float16 tile[64][64 + 8];   // 72-half stride (144B, 16B mult)
    const int tid = threadIdx.x;
    const int k0 = blockIdx.y * 64, n0 = blockIdx.x * 64;

    #pragma unroll
    for (int j = 0; j < 4; ++j) {
        int c  = tid + 256 * j;
        int r  = c >> 4;
        int co = (c & 15) << 2;
        const float4 v = *(const float4*)(W + (size_t)(k0 + r) * N + (n0 + co));
        *(v4h*)&tile[r][co] = pk4(v);
    }
    __syncthreads();
    #pragma unroll
    for (int j = 0; j < 4; ++j) {
        int c  = tid + 256 * j;
        int r  = c >> 4;                     // n within tile
        int co = (c & 15) << 2;              // k within tile
        v4h o;
        #pragma unroll
        for (int jj = 0; jj < 4; ++jj) o[jj] = tile[co + jj][r];
        *(v4h*)(Wt + (size_t)(n0 + r) * K + (k0 + co)) = o;
    }
}

// ---------------------------------------------------------------------------
// GEMM:  C[M,N] = act( A[M,K] * W[K,N] + bias[N] )
// A: f16 row-major.  W pre-transposed as Wt[N][K] f16.  C: f16.
// Block 256 thr (8 waves), tile 256(M) x 64(N), K-step 32.
// Each wave: 32 rows -> 2 A-frags x 4 B-frags = 8 WMMAs / K-step.
// Staging = pure b128 copies -> async global->LDS path.
// act: 0 = none, 1 = exact-erf GELU.
// ---------------------------------------------------------------------------
#define TM 256
#define TN 64
#define TK 32

__global__ __launch_bounds__(256)
void gemm_bias_act(const _Float16* __restrict__ A, const _Float16* __restrict__ Wt,
                   const float* __restrict__ bias, _Float16* __restrict__ C,
                   int M, int N, int K, int act)
{
    __shared__ _Float16 As[TM][TK + 8];     // 40-half stride (80B, 16B mult)
    __shared__ _Float16 Bst[TN][TK + 8];    // N-major: Bst[n][k]

    const int tid  = threadIdx.x;
    const int lane = tid & 31;
    const int wave = tid >> 5;
    const int hi   = lane >> 4;
    const int lo   = lane & 15;
    const int m0   = blockIdx.y * TM;
    const int n0   = blockIdx.x * TN;

    v8f acc[2][4] = {};

    for (int k0 = 0; k0 < K; k0 += TK) {
        // Stage A tile: 256x32 halves; 4 b128 async copies per thread.
        #pragma unroll
        for (int j = 0; j < 4; ++j) {
            int c  = tid + 256 * j;
            int r  = c >> 2;
            int co = (c & 3) << 3;
            copy16_g2l(A + (size_t)(m0 + r) * K + (k0 + co), &As[r][co]);
        }
        // Stage B tile: 64(n) x 32(k) halves; 1 b128 async copy per thread.
        {
            int r  = tid >> 2;
            int co = (tid & 3) << 3;
            copy16_g2l(Wt + (size_t)(n0 + r) * K + (k0 + co), &Bst[r][co]);
        }
        if (k0 + TK < K)
            __builtin_prefetch((const void*)(Wt + (size_t)(n0 + (tid >> 2)) * K + k0 + TK), 0, 1);

        async_wait();
        __syncthreads();

        // A fragments: two 16-row blocks per wave; each = 2 contiguous b128s.
        v16h afr[2];
        #pragma unroll
        for (int mm = 0; mm < 2; ++mm) {
            const int arow = wave * 32 + mm * 16 + lo;
            const v8h a0 = *(const v8h*)&As[arow][hi * 8];
            const v8h a1 = *(const v8h*)&As[arow][16 + hi * 8];
            afr[mm] = cat16(a0, a1);
        }
        // 4 column sub-tiles; B frag = 2 contiguous b128s (K-contiguous).
        #pragma unroll
        for (int nn = 0; nn < 4; ++nn) {
            const int brow = nn * 16 + lo;
            const v8h b0 = *(const v8h*)&Bst[brow][hi * 8];
            const v8h b1 = *(const v8h*)&Bst[brow][16 + hi * 8];
            const v16h bfr = cat16(b0, b1);
            acc[0][nn] = __builtin_amdgcn_wmma_f32_16x16x32_f16(
                false, afr[0], false, bfr, (short)0, acc[0][nn], false, false);
            acc[1][nn] = __builtin_amdgcn_wmma_f32_16x16x32_f16(
                false, afr[1], false, bfr, (short)0, acc[1][nn], false, false);
        }
        __syncthreads();
    }

    // Epilogue: bias + optional GELU, f16 store.
    #pragma unroll
    for (int mm = 0; mm < 2; ++mm)
        #pragma unroll
        for (int nn = 0; nn < 4; ++nn) {
            const int col = n0 + nn * 16 + lo;
            const float bv = bias[col];
            #pragma unroll
            for (int r = 0; r < 8; ++r) {
                float v = acc[mm][nn][r] + bv;
                if (act == 1)
                    v = 0.5f * v * (1.0f + erff(v * 0.70710678118654752f));
                const int row = m0 + wave * 32 + mm * 16 + (hi << 3) + r;
                C[(size_t)row * N + col] = (_Float16)v;
            }
        }
}

// ---------------------------------------------------------------------------
// Flash-style attention. Q,K,V: (B*S, D) f16, head h in cols [h*64, h*64+64).
// Block = 128 threads (4 waves); each wave owns one 16-row query tile.
// grid = (S/64, H, B). Out: (B*S, D) f16 (heads re-merged).
// ---------------------------------------------------------------------------
__global__ __launch_bounds__(128)
void attn_kernel(const _Float16* __restrict__ Q, const _Float16* __restrict__ Kb,
                 const _Float16* __restrict__ Vb, const int* __restrict__ mask,
                 _Float16* __restrict__ Out)
{
    const int S = 2048, D = 1024;
    __shared__ _Float16 Ks[32][64 + 8];      // [key][dim]  (72-half stride)
    __shared__ _Float16 Vs_t[64][32 + 8];    // [dim][key]  (40-half stride)
    __shared__ _Float16 Ps[4][16][32 + 8];   // per-wave P  [row][key]

    const int tid  = threadIdx.x;
    const int lane = tid & 31;
    const int wave = tid >> 5;
    const int hi   = lane >> 4;
    const int lo   = lane & 15;
    const int b    = blockIdx.z;
    const int h    = blockIdx.y;
    const int q0   = blockIdx.x * 64 + wave * 16;
    const size_t headBase = (size_t)b * S * D + (size_t)h * 64;

    // Persistent Q fragments: 16 rows x 64 dims = 2 A-frags; b128 loads.
    v16h aq[2];
    #pragma unroll
    for (int c = 0; c < 2; ++c) {
        const _Float16* qrow = Q + headBase + (size_t)(q0 + lo) * D + c * 32;
        const v8h f0 = *(const v8h*)(qrow + hi * 8);
        const v8h f1 = *(const v8h*)(qrow + 16 + hi * 8);
        aq[c] = cat16(f0, f1);
    }

    v8f o[4] = {};
    float rm[8], rl[8];
    #pragma unroll
    for (int r = 0; r < 8; ++r) { rm[r] = -3.0e38f; rl[r] = 0.0f; }

    for (int t = 0; t < S / 32; ++t) {
        const int kRow0 = t * 32;
        // Stage K row-major: 32x64 halves; 2 async b128 copies per thread.
        #pragma unroll
        for (int j = 0; j < 2; ++j) {
            int c  = tid + 128 * j;
            int r  = c >> 3;
            int co = (c & 7) << 3;
            copy16_g2l(Kb + headBase + (size_t)(kRow0 + r) * D + co, &Ks[r][co]);
        }
        // Stage V transposed: thread owns key pair (2kp,2kp+1), 8 dims.
        {
            const int kp = tid >> 3;             // 0..15
            const int co = (tid & 7) << 3;       // dim chunk
            const v8h va = *(const v8h*)(Vb + headBase + (size_t)(kRow0 + 2 * kp) * D + co);
            const v8h vb2 = *(const v8h*)(Vb + headBase + (size_t)(kRow0 + 2 * kp + 1) * D + co);
            #pragma unroll
            for (int j = 0; j < 8; ++j) {
                v2h pr; pr[0] = va[j]; pr[1] = vb2[j];
                *(v2h*)&Vs_t[co + j][2 * kp] = pr;       // b32 store
            }
        }
        async_wait();
        __syncthreads();

        // Scores: S(16x32) = Q(16x64) * K^T ; B-frags K-contiguous -> b128.
        v8f s0 = {}, s1 = {};
        #pragma unroll
        for (int c = 0; c < 2; ++c) {
            const v8h k00 = *(const v8h*)&Ks[lo][c * 32 + hi * 8];
            const v8h k01 = *(const v8h*)&Ks[lo][c * 32 + 16 + hi * 8];
            const v8h k10 = *(const v8h*)&Ks[16 + lo][c * 32 + hi * 8];
            const v8h k11 = *(const v8h*)&Ks[16 + lo][c * 32 + 16 + hi * 8];
            s0 = __builtin_amdgcn_wmma_f32_16x16x32_f16(
                false, aq[c], false, cat16(k00, k01), (short)0, s0, false, false);
            s1 = __builtin_amdgcn_wmma_f32_16x16x32_f16(
                false, aq[c], false, cat16(k10, k11), (short)0, s1, false, false);
        }

        // Scale + additive mask for this lane's two key columns.
        const float mb0 = -10000.0f * (1.0f - (float)mask[b * S + kRow0 + lo]);
        const float mb1 = -10000.0f * (1.0f - (float)mask[b * S + kRow0 + 16 + lo]);
        float mx[8], p0[8], p1[8];
        #pragma unroll
        for (int r = 0; r < 8; ++r) {
            float a0 = s0[r] * 0.125f + mb0;
            float a1 = s1[r] * 0.125f + mb1;
            s0[r] = a0; s1[r] = a1;
            mx[r] = fmaxf(a0, a1);
        }
        #pragma unroll
        for (int off = 1; off <= 8; off <<= 1)
            #pragma unroll
            for (int r = 0; r < 8; ++r)
                mx[r] = fmaxf(mx[r], __shfl_xor(mx[r], off, 32));

        float scl[8];
        #pragma unroll
        for (int r = 0; r < 8; ++r) {
            const float nm = fmaxf(rm[r], mx[r]);
            scl[r] = expf(rm[r] - nm);
            rm[r] = nm;
            p0[r] = expf(s0[r] - nm);
            p1[r] = expf(s1[r] - nm);
        }
        float rs[8];
        #pragma unroll
        for (int r = 0; r < 8; ++r) rs[r] = p0[r] + p1[r];
        #pragma unroll
        for (int off = 1; off <= 8; off <<= 1)
            #pragma unroll
            for (int r = 0; r < 8; ++r)
                rs[r] += __shfl_xor(rs[r], off, 32);
        #pragma unroll
        for (int r = 0; r < 8; ++r) rl[r] = rl[r] * scl[r] + rs[r];
        #pragma unroll
        for (int nn = 0; nn < 4; ++nn)
            #pragma unroll
            for (int r = 0; r < 8; ++r)
                o[nn][r] *= scl[r];

        // P (C-layout) -> LDS -> A-layout fragment (b128 reads).
        #pragma unroll
        for (int r = 0; r < 8; ++r) {
            Ps[wave][(hi << 3) + r][lo]      = (_Float16)p0[r];
            Ps[wave][(hi << 3) + r][16 + lo] = (_Float16)p1[r];
        }
        __syncthreads();
        const v8h pa0 = *(const v8h*)&Ps[wave][lo][hi * 8];
        const v8h pa1 = *(const v8h*)&Ps[wave][lo][16 + hi * 8];
        const v16h pa = cat16(pa0, pa1);

        // o(16x64) += P(16x32) * V(32x64); V frags from transposed tile.
        #pragma unroll
        for (int nn = 0; nn < 4; ++nn) {
            const int vrow = nn * 16 + lo;
            const v8h b0 = *(const v8h*)&Vs_t[vrow][hi * 8];
            const v8h b1 = *(const v8h*)&Vs_t[vrow][16 + hi * 8];
            o[nn] = __builtin_amdgcn_wmma_f32_16x16x32_f16(
                false, pa, false, cat16(b0, b1), (short)0, o[nn], false, false);
        }
        __syncthreads();
    }

    // Normalize and write back (heads merged, f16).
    #pragma unroll
    for (int nn = 0; nn < 4; ++nn)
        #pragma unroll
        for (int r = 0; r < 8; ++r) {
            const int row = q0 + (hi << 3) + r;
            Out[headBase + (size_t)row * D + nn * 16 + lo] = (_Float16)(o[nn][r] / rl[r]);
        }
}

// ---------------------------------------------------------------------------
// Residual + LayerNorm: Out[row,:] = g * norm(X[row,:] + R[row,:]) + b
// X from either fp32 (Xf) or f16 (Xh) - exactly one non-null.
// Output to f16 (OutH) or fp32 (OutF) - exactly one non-null.
// One block (256 threads) per row, D = 1024.
// ---------------------------------------------------------------------------
__global__ __launch_bounds__(256)
void add_ln_kernel(const float* __restrict__ Xf, const _Float16* __restrict__ Xh,
                   const _Float16* __restrict__ R,
                   const float* __restrict__ g, const float* __restrict__ bta,
                   _Float16* __restrict__ OutH, float* __restrict__ OutF)
{
    const int D = 1024;
    __shared__ float sh[8];
    const int row  = blockIdx.x;
    const int tid  = threadIdx.x;
    const int lane = tid & 31;
    const int wave = tid >> 5;
    const int col  = tid << 2;

    float v[4];
    {
        const v4h ra = *(const v4h*)(R + (size_t)row * D + col);
        if (Xf) {
            const float4 xa = *(const float4*)(Xf + (size_t)row * D + col);
            v[0] = xa.x + (float)ra[0]; v[1] = xa.y + (float)ra[1];
            v[2] = xa.z + (float)ra[2]; v[3] = xa.w + (float)ra[3];
        } else {
            const v4h xa = *(const v4h*)(Xh + (size_t)row * D + col);
            #pragma unroll
            for (int j = 0; j < 4; ++j) v[j] = (float)xa[j] + (float)ra[j];
        }
    }
    float s = v[0] + v[1] + v[2] + v[3];

    #pragma unroll
    for (int off = 16; off >= 1; off >>= 1) s += __shfl_xor(s, off, 32);
    if (lane == 0) sh[wave] = s;
    __syncthreads();
    float tot = 0.0f;
    #pragma unroll
    for (int w = 0; w < 8; ++w) tot += sh[w];
    const float mean = tot * (1.0f / 1024.0f);
    __syncthreads();

    float d = 0.0f;
    #pragma unroll
    for (int j = 0; j < 4; ++j) { const float t = v[j] - mean; d += t * t; }
    #pragma unroll
    for (int off = 16; off >= 1; off >>= 1) d += __shfl_xor(d, off, 32);
    if (lane == 0) sh[wave] = d;
    __syncthreads();
    float totd = 0.0f;
    #pragma unroll
    for (int w = 0; w < 8; ++w) totd += sh[w];
    const float inv = rsqrtf(totd * (1.0f / 1024.0f) + 1e-12f);

    const float4 gg = *(const float4*)(g + col);
    const float4 bb = *(const float4*)(bta + col);
    float4 ov;
    ov.x = gg.x * (v[0] - mean) * inv + bb.x;
    ov.y = gg.y * (v[1] - mean) * inv + bb.y;
    ov.z = gg.z * (v[2] - mean) * inv + bb.z;
    ov.w = gg.w * (v[3] - mean) * inv + bb.w;
    if (OutF) {
        *(float4*)(OutF + (size_t)row * D + col) = ov;
    } else {
        *(v4h*)(OutH + (size_t)row * D + col) = pk4(ov);
    }
}

// ---------------------------------------------------------------------------
// Launch: full encoder layer. B=2, S=2048, D=1024, H=16, F=4096.
// Workspace (f16 activations), offsets in MB:
//   [0,8)    x16          [8,16)  Q -> PROJ     [16,24) K -> H1
//   [24,32)  V -> FF2     [32,40) ATT           [40,72) FF1
//   [72,96)  f16 transposed weights (Q,K,V,O 2MB each; Wt1,Wt2 8MB)
// Total required: 96 MB.
// ---------------------------------------------------------------------------
extern "C" void kernel_launch(void* const* d_in, const int* in_sizes, int n_in,
                              void* d_out, int out_size, void* d_ws, size_t ws_size,
                              hipStream_t stream) {
    const float* x   = (const float*)d_in[0];
    const int*   msk = (const int*)  d_in[1];
    const float* wq  = (const float*)d_in[2];
    const float* bq  = (const float*)d_in[3];
    const float* wk  = (const float*)d_in[4];
    const float* bk  = (const float*)d_in[5];
    const float* wv  = (const float*)d_in[6];
    const float* bv  = (const float*)d_in[7];
    const float* wo  = (const float*)d_in[8];
    const float* bo  = (const float*)d_in[9];
    const float* g1  = (const float*)d_in[10];
    const float* b1  = (const float*)d_in[11];
    const float* w1  = (const float*)d_in[12];
    const float* bf1 = (const float*)d_in[13];
    const float* w2  = (const float*)d_in[14];
    const float* bf2 = (const float*)d_in[15];
    const float* g2  = (const float*)d_in[16];
    const float* b2  = (const float*)d_in[17];

    const int BS = 2 * 2048, D = 1024, F = 4096;
    const size_t MB = 1024ull * 1024ull;

    char* ws = (char*)d_ws;
    _Float16* X16  = (_Float16*)(ws + 0 * MB);
    _Float16* Qb   = (_Float16*)(ws + 8 * MB);
    _Float16* Kbuf = (_Float16*)(ws + 16 * MB);
    _Float16* Vbuf = (_Float16*)(ws + 24 * MB);
    _Float16* ATT  = (_Float16*)(ws + 32 * MB);
    _Float16* FF1  = (_Float16*)(ws + 40 * MB);
    _Float16* PROJ = Qb;                     // reuse after attention
    _Float16* H1   = Kbuf;                   // reuse after attention
    _Float16* FF2  = Vbuf;                   // reuse after attention

    _Float16* WtQ = (_Float16*)(ws + 72 * MB);
    _Float16* WtK = (_Float16*)(ws + 74 * MB);
    _Float16* WtV = (_Float16*)(ws + 76 * MB);
    _Float16* WtO = (_Float16*)(ws + 78 * MB);
    _Float16* Wt1 = (_Float16*)(ws + 80 * MB);   // [4096][1024]
    _Float16* Wt2 = (_Float16*)(ws + 88 * MB);   // [1024][4096]

    const dim3 blk(256);

    // Input convert + weight transpose/convert
    f32to16_kernel<<<dim3((BS * D) / 1024), blk, 0, stream>>>(x, X16);
    wtrans_kernel<<<dim3(D / 64, D / 64), blk, 0, stream>>>(wq, WtQ, D, D);
    wtrans_kernel<<<dim3(D / 64, D / 64), blk, 0, stream>>>(wk, WtK, D, D);
    wtrans_kernel<<<dim3(D / 64, D / 64), blk, 0, stream>>>(wv, WtV, D, D);
    wtrans_kernel<<<dim3(D / 64, D / 64), blk, 0, stream>>>(wo, WtO, D, D);
    wtrans_kernel<<<dim3(F / 64, D / 64), blk, 0, stream>>>(w1, Wt1, D, F);
    wtrans_kernel<<<dim3(D / 64, F / 64), blk, 0, stream>>>(w2, Wt2, F, D);

    const dim3 gridD(D / TN, BS / TM);   // N=1024 GEMMs
    const dim3 gridF(F / TN, BS / TM);   // N=4096 GEMM

    // QKV projections
    gemm_bias_act<<<gridD, blk, 0, stream>>>(X16, WtQ, bq, Qb,   BS, D, D, 0);
    gemm_bias_act<<<gridD, blk, 0, stream>>>(X16, WtK, bk, Kbuf, BS, D, D, 0);
    gemm_bias_act<<<gridD, blk, 0, stream>>>(X16, WtV, bv, Vbuf, BS, D, D, 0);

    // Attention (flash-style, heads re-merged into ATT)
    attn_kernel<<<dim3(2048 / 64, 16, 2), dim3(128), 0, stream>>>(Qb, Kbuf, Vbuf, msk, ATT);

    // Output projection
    gemm_bias_act<<<gridD, blk, 0, stream>>>(ATT, WtO, bo, PROJ, BS, D, D, 0);

    // Residual + LN1 (residual from original fp32 x)
    add_ln_kernel<<<dim3(BS), blk, 0, stream>>>(x, nullptr, PROJ, g1, b1, H1, nullptr);

    // FFN
    gemm_bias_act<<<gridF, blk, 0, stream>>>(H1, Wt1, bf1, FF1, BS, F, D, 1);
    gemm_bias_act<<<gridD, blk, 0, stream>>>(FF1, Wt2, bf2, FF2, BS, D, F, 0);

    // Residual + LN2 -> fp32 output
    add_ln_kernel<<<dim3(BS), blk, 0, stream>>>(nullptr, H1, FF2, g2, b2, nullptr, (float*)d_out);
}
